// BahdanauAttention_28965259444466
// MI455X (gfx1250) — compile-verified, matrix operand order checked by reading, not run on hardware
//
#include <hip/hip_runtime.h>
#include <hip/hip_bf16.h>

typedef __attribute__((ext_vector_type(16))) __bf16 v16bf;
typedef __attribute__((ext_vector_type(8)))  __bf16 v8bf;
typedef __attribute__((ext_vector_type(8)))  float  v8f;

#define DD 512
#define TT 2048
#define BB 64
#define SPLITS 16
#define TSP (TT / SPLITS)   // 128 timesteps per split
#define ROWS 32             // rows (timesteps) per big tile
#define TILES (TSP / ROWS)  // 4
#define NT 4                // 16-wide N tiles per wave
#define HPAD 8              // bf16 row padding (row stride 520 -> conflict-free frag loads)

// ---------------------------------------------------------------------------
// Kernel 0a: swizzle Wh (f32 [D][D], row = output feature e, col = input d)
// into bf16 B-operand fragments. Layout: frag element
//   whbf[(((kstep*32 + ntile)*32 + lane)*16) + idx]
// matching the 16-bit 32x16 B-matrix per-lane VGPR layout:
//   N = ntile*16 + (lane&15)
//   K = kstep*32 + (lane<16 ? 0 : 8) + (idx&7) + (idx>=8 ? 16 : 0)
// and B[k][n] = Wh[n][k]  (GEMM computes enc[t,e] = sum_d h[t,d]*Wh[e,d]).
// ---------------------------------------------------------------------------
__global__ __launch_bounds__(256) void prep_wh(const float* __restrict__ Wh,
                                               __bf16* __restrict__ whbf) {
  int o = blockIdx.x * 256 + threadIdx.x;  // 0 .. 262143
  int idx   = o & 15;
  int lane  = (o >> 4) & 31;
  int ntile = (o >> 9) & 31;
  int kstep = o >> 14;
  int hi    = (idx >> 3) & 1;
  int base  = (lane < 16) ? 0 : 8;
  int K = kstep * 32 + base + (idx & 7) + hi * 16;
  int N = ntile * 16 + (lane & 15);
  whbf[o] = (__bf16)Wh[N * DD + K];
}

// ---------------------------------------------------------------------------
// Kernel 0b: dec[b][d] = s_t[b] . Ws_w[d] + Ws_b[d]   (tiny GEMV, f32)
// ---------------------------------------------------------------------------
__global__ __launch_bounds__(256) void prep_dec(const float* __restrict__ s_t,
                                                const float* __restrict__ Ws_w,
                                                const float* __restrict__ Ws_b,
                                                float* __restrict__ dec) {
  int o = blockIdx.x * 256 + threadIdx.x;  // 0 .. 32767
  int b = o >> 9, d = o & 511;
  const float4* sp = (const float4*)(s_t + (size_t)b * 1024);
  const float4* wp = (const float4*)(Ws_w + (size_t)d * 1024);
  float s = Ws_b[d];
  for (int k = 0; k < 256; ++k) {
    float4 a = sp[k], w = wp[k];
    s += a.x * w.x + a.y * w.y + a.z * w.z + a.w * w.w;
  }
  dec[o] = s;
}

// ---------------------------------------------------------------------------
// Kernel 1: fused scores GEMM (bf16 WMMA) + tanh + dot-V + online softmax +
// context accumulation. One pass over h_i. Grid = B * SPLITS, 256 threads.
// ---------------------------------------------------------------------------
__global__ __launch_bounds__(256) void attn_main(
    const float* __restrict__ h_i, const float* __restrict__ coverage,
    const float* __restrict__ dec, const __bf16* __restrict__ whbf,
    const float* __restrict__ Wc, const float* __restrict__ Vw,
    float* __restrict__ e_buf, float* __restrict__ stats,
    float* __restrict__ ctxp) {
  __shared__ __bf16 hbf[ROWS][DD + HPAD];   // bf16 h tile (A operand + context src)
  __shared__ float  e_sm[ROWS];
  __shared__ float  cov_sm[ROWS];
  __shared__ float  w_sm[ROWS];
  __shared__ float  bc_sm[1];               // broadcast rescale factor

  const int tid   = threadIdx.x;
  const int wave  = tid >> 5;
  const int lane  = tid & 31;
  const int l16   = lane & 15;
  const int hsel  = lane >> 4;
  const int b     = blockIdx.x / SPLITS;
  const int split = blockIdx.x - b * SPLITS;
  const int t0    = split * TSP;

  // Per-wave column constants (each wave owns 64 output features).
  float decv[NT], wcv[NT], vv[NT];
#pragma unroll
  for (int nt = 0; nt < NT; ++nt) {
    int col = wave * (NT * 16) + nt * 16 + l16;
    decv[nt] = dec[b * DD + col];
    wcv[nt]  = Wc[col];
    vv[nt]   = Vw[col];
  }

  // Per-wave B-fragment base: element offset = ks*16384 + nt*512 from here.
  const __bf16* wbase = whbf + (size_t)(wave * NT * 32 + lane) * 16;
  // A-fragment LDS bases (per-ks offsets are ds immediate offsets).
  const __bf16* abase0 = &hbf[l16][hsel * 8];
  const __bf16* abase1 = &hbf[16 + l16][hsel * 8];

  float m_run = -3.0e38f, l_run = 0.f;  // meaningful in tid 0 only
  float c0 = 0.f, c1 = 0.f;             // context accum: cols tid, tid+256
  const v8f vzero = {0.f, 0.f, 0.f, 0.f, 0.f, 0.f, 0.f, 0.f};

  for (int tt = 0; tt < TILES; ++tt) {
    const int tr = t0 + tt * ROWS;
    if (tid < ROWS) {
      e_sm[tid]   = 0.f;
      cov_sm[tid] = coverage[(size_t)b * TT + tr + tid];
    }
    // Stage h tile: ROWS x DD f32 from HBM -> bf16 LDS (4096 float4 loads).
    const float4* hsrc = (const float4*)(h_i + ((size_t)b * TT + tr) * DD);
#pragma unroll
    for (int j = 0; j < 16; ++j) {
      int f4  = j * 256 + tid;
      int row = f4 >> 7;            // 128 float4 per row
      int c4  = (f4 & 127) << 2;
      float4 v = hsrc[f4];
      hbf[row][c4 + 0] = (__bf16)v.x;
      hbf[row][c4 + 1] = (__bf16)v.y;
      hbf[row][c4 + 2] = (__bf16)v.z;
      hbf[row][c4 + 3] = (__bf16)v.w;
      if (tt + 1 < TILES)
        __builtin_prefetch((const void*)(hsrc + (size_t)ROWS * 128 + f4), 0, 0);
    }
    __syncthreads();

    // --- GEMM: enc[32 x 512] = h_tile @ Wh^T (bf16 WMMA, f32 accum) ---
    // Two-buffer ping-pong on B fragments, #pragma unroll 1 so peak VGPR
    // pressure is structurally capped (acc 64 + 2x32 frag regs): no spills,
    // and every WMMA consumes loads issued one full K-step earlier.
    v8f acc[2][NT];
#pragma unroll
    for (int mt = 0; mt < 2; ++mt)
#pragma unroll
      for (int nt = 0; nt < NT; ++nt) acc[mt][nt] = vzero;

    v16bf bufA[NT], bufB[NT];
#pragma unroll
    for (int nt = 0; nt < NT; ++nt)
      bufA[nt] = *(const v16bf*)(wbase + nt * 512);

    const __bf16* wptr = wbase + 16384;   // K-step ks+1 base
    const __bf16* aptr0 = abase0;
    const __bf16* aptr1 = abase1;
#pragma unroll 1
    for (int ks = 0; ks < 16; ks += 2) {
      // prefetch ks+1 -> bufB
#pragma unroll
      for (int nt = 0; nt < NT; ++nt)
        bufB[nt] = *(const v16bf*)(wptr + nt * 512);
      union { v16bf v; v8bf h[2]; } a0, a1;
      a0.h[0] = *(const v8bf*)(aptr0);
      a0.h[1] = *(const v8bf*)(aptr0 + 16);
      a1.h[0] = *(const v8bf*)(aptr1);
      a1.h[1] = *(const v8bf*)(aptr1 + 16);
#pragma unroll
      for (int nt = 0; nt < NT; ++nt) {
        acc[0][nt] = __builtin_amdgcn_wmma_f32_16x16x32_bf16(
            false, a0.v, false, bufA[nt], (short)0, acc[0][nt], false, false);
        acc[1][nt] = __builtin_amdgcn_wmma_f32_16x16x32_bf16(
            false, a1.v, false, bufA[nt], (short)0, acc[1][nt], false, false);
      }
      // prefetch ks+2 -> bufA (last iteration harmlessly over-reads 32KB into
      // the adjacent dec workspace region; data unused)
#pragma unroll
      for (int nt = 0; nt < NT; ++nt)
        bufA[nt] = *(const v16bf*)(wptr + 16384 + nt * 512);
      a0.h[0] = *(const v8bf*)(aptr0 + 32);
      a0.h[1] = *(const v8bf*)(aptr0 + 48);
      a1.h[0] = *(const v8bf*)(aptr1 + 32);
      a1.h[1] = *(const v8bf*)(aptr1 + 48);
#pragma unroll
      for (int nt = 0; nt < NT; ++nt) {
        acc[0][nt] = __builtin_amdgcn_wmma_f32_16x16x32_bf16(
            false, a0.v, false, bufB[nt], (short)0, acc[0][nt], false, false);
        acc[1][nt] = __builtin_amdgcn_wmma_f32_16x16x32_bf16(
            false, a1.v, false, bufB[nt], (short)0, acc[1][nt], false, false);
      }
      wptr  += 32768;   // two K-steps
      aptr0 += 64;
      aptr1 += 64;
    }

    // --- e_t = V . tanh(enc + dec + cov*Wc), reduced over D ---
#pragma unroll
    for (int mt = 0; mt < 2; ++mt) {
#pragma unroll
      for (int r = 0; r < 8; ++r) {
        int row = mt * 16 + r + hsel * 8;   // C layout: lanes>=16 hold M=r+8
        float covt = cov_sm[row];
        float p = 0.f;
#pragma unroll
        for (int nt = 0; nt < NT; ++nt) {
          float f = acc[mt][nt][r] + decv[nt] + covt * wcv[nt];
          p += tanhf(f) * vv[nt];
        }
        p += __shfl_xor(p, 1, 32);
        p += __shfl_xor(p, 2, 32);
        p += __shfl_xor(p, 4, 32);
        p += __shfl_xor(p, 8, 32);
        if (l16 == 0) atomicAdd(&e_sm[row], p);   // ds_add_f32 across waves
      }
    }
    __syncthreads();

    // --- online softmax bookkeeping (thread 0) ---
    if (tid == 0) {
      float tmax = -3.0e38f;
      for (int t = 0; t < ROWS; ++t) tmax = fmaxf(tmax, e_sm[t]);
      float mnew = fmaxf(m_run, tmax);
      float r = __expf(m_run - mnew);
      float ls = 0.f;
      for (int t = 0; t < ROWS; ++t) {
        float w = __expf(e_sm[t] - mnew);
        w_sm[t] = w;
        ls += w;
      }
      l_run = l_run * r + ls;
      m_run = mnew;
      bc_sm[0] = r;
    }
    if (tid < ROWS) e_buf[(size_t)b * TT + tr + tid] = e_sm[tid];  // raw score
    __syncthreads();

    // --- rescale + accumulate context from the LDS-resident tile ---
    float r = bc_sm[0];
    c0 *= r;
    c1 *= r;
#pragma unroll 8
    for (int t = 0; t < ROWS; ++t) {
      float w = w_sm[t];
      c0 += w * (float)hbf[t][tid];
      c1 += w * (float)hbf[t][tid + 256];
    }
    __syncthreads();
  }

  if (tid == 0) {
    stats[(size_t)(b * SPLITS + split) * 2 + 0] = m_run;
    stats[(size_t)(b * SPLITS + split) * 2 + 1] = l_run;
  }
  float* cp = ctxp + (size_t)(b * SPLITS + split) * DD;
  cp[tid]       = c0;
  cp[tid + 256] = c1;
}

// ---------------------------------------------------------------------------
// Kernel 2: logsumexp-merge split stats, combine context partials, emit
// a_t and new_coverage. Grid = B, 256 threads.
// ---------------------------------------------------------------------------
__global__ __launch_bounds__(256) void attn_finalize(
    const float* __restrict__ stats, const float* __restrict__ ctxp,
    const float* __restrict__ e_buf, const float* __restrict__ coverage,
    float* __restrict__ out) {
  const int b = blockIdx.x, tid = threadIdx.x;
  float M = -3.0e38f;
#pragma unroll
  for (int k = 0; k < SPLITS; ++k) M = fmaxf(M, stats[(b * SPLITS + k) * 2]);
  float L = 0.f, scale[SPLITS];
#pragma unroll
  for (int k = 0; k < SPLITS; ++k) {
    scale[k] = __expf(stats[(b * SPLITS + k) * 2] - M);
    L += stats[(b * SPLITS + k) * 2 + 1] * scale[k];
  }
  float invL = 1.f / L;
  float s0 = 0.f, s1 = 0.f;
#pragma unroll
  for (int k = 0; k < SPLITS; ++k) {
    const float* cp = ctxp + (size_t)(b * SPLITS + k) * DD;
    s0 += cp[tid] * scale[k];
    s1 += cp[tid + 256] * scale[k];
  }
  out[b * DD + tid]       = s0 * invL;
  out[b * DD + tid + 256] = s1 * invL;

  float* a_out = out + BB * DD;            // a_t   [B,T]
  float* c_out = out + BB * DD + BB * TT;  // new_coverage [B,T]
  for (int j = 0; j < TT / 256; ++j) {
    size_t ix = (size_t)b * TT + j * 256 + tid;
    float a = __expf(e_buf[ix] - M) * invL;
    a_out[ix] = a;
    c_out[ix] = coverage[ix] + a;
  }
}

// ---------------------------------------------------------------------------
extern "C" void kernel_launch(void* const* d_in, const int* in_sizes, int n_in,
                              void* d_out, int out_size, void* d_ws,
                              size_t ws_size, hipStream_t stream) {
  const float* h_i = (const float*)d_in[0];
  const float* s_t = (const float*)d_in[1];
  const float* cov = (const float*)d_in[2];
  const float* Wh  = (const float*)d_in[3];
  const float* Wsw = (const float*)d_in[4];
  const float* Wsb = (const float*)d_in[5];
  const float* Wc  = (const float*)d_in[6];
  const float* Vw  = (const float*)d_in[7];
  float* out = (float*)d_out;

  char* ws = (char*)d_ws;
  size_t off = 0;
  __bf16* whbf = (__bf16*)(ws + off); off += (size_t)512 << 10;  // 512 KB swizzled Wh
  float* dec   = (float*)(ws + off);  off += (size_t)128 << 10;  // 128 KB dec features
  float* e_buf = (float*)(ws + off);  off += (size_t)512 << 10;  // 512 KB raw scores
  float* stats = (float*)(ws + off);  off += (size_t)8 << 10;    //   8 KB (m,l) per split
  float* ctxp  = (float*)(ws + off);                             //   2 MB context partials

  prep_wh<<<1024, 256, 0, stream>>>(Wh, whbf);
  prep_dec<<<128, 256, 0, stream>>>(s_t, Wsw, Wsb, dec);
  attn_main<<<BB * SPLITS, 256, 0, stream>>>(h_i, cov, dec, whbf, Wc, Vw,
                                             e_buf, stats, ctxp);
  attn_finalize<<<BB, 256, 0, stream>>>(stats, ctxp, e_buf, cov, out);
}
